// MultiHeadMlp_66099546685507
// MI455X (gfx1250) — compile-verified, matrix operand order checked by reading, not run on hardware
//
#include <hip/hip_runtime.h>
#include <stdint.h>

// ---------------- problem constants ----------------
#define BATCH     8
#define NTOK      4096
#define DIM       1024
#define NUM_HEADS 4
#define HEAD_DIM  256
#define HIDDEN    1024
#define SQUEEZE   64

#define TOK_TILE  32                 // tokens per workgroup (divides NTOK)
#define XS_LD     (HEAD_DIM + 8)     // padded LDS leading dims (even!)
#define HS_LD     (HIDDEN + 8)
#define XS_BYTES  (TOK_TILE * XS_LD * 2)
#define HS_BYTES  (TOK_TILE * HS_LD * 2)

// workspace layout (bytes)
#define W1T_BYTES (NUM_HEADS * HIDDEN * HEAD_DIM * 2)   // bf16 [h][e][d]
#define W2T_BYTES (NUM_HEADS * HEAD_DIM * HIDDEN * 2)   // bf16 [h][n][k]
#define POOLED_OFF (W1T_BYTES + W2T_BYTES)
#define GATE_OFF   (POOLED_OFF + BATCH * DIM * 4)

typedef __attribute__((ext_vector_type(16))) __bf16 v16bf;
typedef __attribute__((ext_vector_type(8)))  float  v8f;

// f32 -> bf16 bits, round-to-nearest-even
__device__ __forceinline__ unsigned short f2bf_bits(float f) {
  union { float f; uint32_t u; } a; a.f = f;
  uint32_t u = a.u;
  uint32_t r = (u + 0x7FFFu + ((u >> 16) & 1u)) >> 16;
  return (unsigned short)r;
}

// Build a 16x32 bf16 WMMA fragment from a row-major bf16 buffer.
// A-matrix layout (ISA 7.12.2, wave32): lanes 0-15 -> M=lane, K {0..7,16..23};
// lanes 16-31 -> M=lane-16, K {8..15,24..31}. B uses the mirrored layout with
// the weight matrix stored transposed ([col][k]) so K-pairs are contiguous.
__device__ __forceinline__ v16bf load_frag(const unsigned short* buf, int ld,
                                           int rowBase, int k0, int lane) {
  int row = rowBase + (lane & 15);
  int kb  = k0 + ((lane >> 4) << 3);            // even
  const uint32_t* src = (const uint32_t*)(buf + (size_t)row * ld);
  union { uint32_t u[8]; v16bf v; } r;
#pragma unroll
  for (int p = 0; p < 4; ++p) {
    r.u[p]     = src[(kb >> 1) + p];            // K = kb .. kb+7
    r.u[4 + p] = src[((kb + 16) >> 1) + p];     // K = kb+16 .. kb+23
  }
  return r.v;
}

// ---------------- phase 0: weight conversion / zeroing ----------------
__global__ void convert_w1_kernel(const float* __restrict__ W1, unsigned short* __restrict__ w1t) {
  int idx = blockIdx.x * 256 + threadIdx.x;               // over 4*256*1024
  if (idx >= NUM_HEADS * HEAD_DIM * HIDDEN) return;
  int e = idx & (HIDDEN - 1);
  int d = (idx / HIDDEN) & (HEAD_DIM - 1);
  int h = idx / (HIDDEN * HEAD_DIM);
  w1t[((size_t)(h * HIDDEN + e)) * HEAD_DIM + d] = f2bf_bits(W1[idx]);
}

__global__ void convert_w2_kernel(const float* __restrict__ W2, unsigned short* __restrict__ w2t) {
  int idx = blockIdx.x * 256 + threadIdx.x;               // over 4*1024*256
  if (idx >= NUM_HEADS * HIDDEN * HEAD_DIM) return;
  int n = idx & (HEAD_DIM - 1);
  int k = (idx / HEAD_DIM) & (HIDDEN - 1);
  int h = idx / (HIDDEN * HEAD_DIM);
  w2t[((size_t)(h * HEAD_DIM + n)) * HIDDEN + k] = f2bf_bits(W2[idx]);
}

__global__ void zero_f32_kernel(float* __restrict__ p, int n) {
  int idx = blockIdx.x * 256 + threadIdx.x;
  if (idx < n) p[idx] = 0.0f;
}

// ---------------- phase 1: fused per-head MLP + pooled partial sums ----------------
__global__ __launch_bounds__(256)
void mlp_wmma_kernel(const float* __restrict__ x,
                     const float* __restrict__ b1,
                     const float* __restrict__ b2,
                     const unsigned short* __restrict__ w1t,
                     const unsigned short* __restrict__ w2t,
                     float* __restrict__ out,
                     float* __restrict__ pooled) {
  extern __shared__ char smem[];
  unsigned short* xs = (unsigned short*)smem;                 // [32][XS_LD] bf16
  unsigned short* hs = (unsigned short*)(smem + XS_BYTES);    // [32][HS_LD] bf16

  const int tile  = blockIdx.x;            // token tile (0..1023)
  const int head  = blockIdx.y;            // 0..3
  const int tid   = threadIdx.x;
  const int wave  = tid >> 5;              // 0..7
  const int lane  = tid & 31;
  const int token0 = tile * TOK_TILE;      // flat token index b*NTOK + n
  const int rowoff = (lane < 16) ? 0 : 8;  // C-matrix row offset per lane half

  // ---- stage x tile (32 tokens x 256 chans of this head) as bf16 ----
  const float* xsrc = x + (size_t)token0 * DIM + head * HEAD_DIM;
#pragma unroll
  for (int r = 0; r < TOK_TILE; ++r)
    xs[r * XS_LD + tid] = f2bf_bits(xsrc[(size_t)r * DIM + tid]);
  __syncthreads();

  // ---- GEMM1: [32 x 256] x [256 x 1024], wave owns 128 hidden cols ----
  const unsigned short* w1h = w1t + (size_t)head * HIDDEN * HEAD_DIM;
  v8f zz = {};
  v8f acc[2][8];
#pragma unroll
  for (int m = 0; m < 2; ++m)
#pragma unroll
    for (int n = 0; n < 8; ++n) acc[m][n] = zz;

  for (int kb = 0; kb < HEAD_DIM; kb += 32) {
    v16bf a0 = load_frag(xs, XS_LD, 0,  kb, lane);
    v16bf a1 = load_frag(xs, XS_LD, 16, kb, lane);
#pragma unroll
    for (int n = 0; n < 8; ++n) {
      v16bf bm = load_frag(w1h, HEAD_DIM, wave * 128 + n * 16, kb, lane);
      acc[0][n] = __builtin_amdgcn_wmma_f32_16x16x32_bf16(false, a0, false, bm,
                                                          (short)0, acc[0][n], false, false);
      acc[1][n] = __builtin_amdgcn_wmma_f32_16x16x32_bf16(false, a1, false, bm,
                                                          (short)0, acc[1][n], false, false);
    }
  }

  // ---- bias + exact GELU -> bf16 hidden tile in LDS ----
#pragma unroll
  for (int m = 0; m < 2; ++m)
#pragma unroll
    for (int n = 0; n < 8; ++n) {
      int col  = wave * 128 + n * 16 + (lane & 15);
      float bias = b1[head * HIDDEN + col];
#pragma unroll
      for (int r = 0; r < 8; ++r) {
        float v = acc[m][n][r] + bias;
        v = 0.5f * v * (1.0f + erff(v * 0.70710678118654752f));
        hs[(m * 16 + rowoff + r) * HS_LD + col] = f2bf_bits(v);
      }
    }
  __syncthreads();

  // ---- GEMM2: [32 x 1024] x [1024 x 256], wave owns 32 out cols ----
  const unsigned short* w2h = w2t + (size_t)head * HEAD_DIM * HIDDEN;
  const int ncb = wave * 32;
  v8f oa[2][2];
  oa[0][0] = zz; oa[0][1] = zz; oa[1][0] = zz; oa[1][1] = zz;

  for (int kb = 0; kb < HIDDEN; kb += 32) {
    v16bf a0 = load_frag(hs, HS_LD, 0,  kb, lane);
    v16bf a1 = load_frag(hs, HS_LD, 16, kb, lane);
    v16bf bm0 = load_frag(w2h, HIDDEN, ncb,      kb, lane);
    v16bf bm1 = load_frag(w2h, HIDDEN, ncb + 16, kb, lane);
    oa[0][0] = __builtin_amdgcn_wmma_f32_16x16x32_bf16(false, a0, false, bm0,
                                                       (short)0, oa[0][0], false, false);
    oa[0][1] = __builtin_amdgcn_wmma_f32_16x16x32_bf16(false, a0, false, bm1,
                                                       (short)0, oa[0][1], false, false);
    oa[1][0] = __builtin_amdgcn_wmma_f32_16x16x32_bf16(false, a1, false, bm0,
                                                       (short)0, oa[1][0], false, false);
    oa[1][1] = __builtin_amdgcn_wmma_f32_16x16x32_bf16(false, a1, false, bm1,
                                                       (short)0, oa[1][1], false, false);
  }

  // ---- bias, store out (f32), accumulate pooled column sums ----
  const int bidx = token0 >> 12;   // token0 / NTOK
#pragma unroll
  for (int m = 0; m < 2; ++m)
#pragma unroll
    for (int n = 0; n < 2; ++n) {
      int colh = ncb + n * 16 + (lane & 15);         // 0..255 within head
      int col  = head * HEAD_DIM + colh;             // 0..1023
      float bias = b2[head * HEAD_DIM + colh];
      float csum = 0.0f;
#pragma unroll
      for (int r = 0; r < 8; ++r) {
        float v = oa[m][n][r] + bias;
        out[(size_t)(token0 + m * 16 + rowoff + r) * DIM + col] = v;
        csum += v;
      }
      atomicAdd(&pooled[bidx * DIM + col], csum);
    }
}

// ---------------- phase 2: SE gate (tiny) ----------------
__global__ __launch_bounds__(256)
void gate_kernel(const float* __restrict__ pooled,
                 const float* __restrict__ cw1, const float* __restrict__ cb1,
                 const float* __restrict__ cw2, const float* __restrict__ cb2,
                 float* __restrict__ gate) {
  __shared__ float pl[DIM];
  __shared__ float sq[SQUEEZE];
  const int b = blockIdx.x;
  const int tid = threadIdx.x;
  for (int c = tid; c < DIM; c += 256)
    pl[c] = pooled[b * DIM + c] * (1.0f / (float)NTOK);
  __syncthreads();
  if (tid < SQUEEZE) {
    float a = cb1[tid];
    for (int c = 0; c < DIM; ++c) a += pl[c] * cw1[c * SQUEEZE + tid];
    sq[tid] = fmaxf(a, 0.0f);
  }
  __syncthreads();
  for (int c = tid; c < DIM; c += 256) {
    float a = cb2[c];
#pragma unroll
    for (int j = 0; j < SQUEEZE; ++j) a += sq[j] * cw2[j * DIM + c];
    gate[b * DIM + c] = 1.0f / (1.0f + __expf(-a));
  }
}

// ---------------- phase 3: out = out * (1 + gate) ----------------
__global__ __launch_bounds__(256)
void apply_gate_kernel(float* __restrict__ out, const float* __restrict__ gate) {
  int idx4 = blockIdx.x * 256 + threadIdx.x;      // one float4 each
  float4* o4 = (float4*)out;
  float4 v = o4[idx4];
  int b = idx4 >> 20;                              // (NTOK*DIM/4) = 1<<20 per batch
  int c = (idx4 & 255) << 2;                       // 256 float4 per row of 1024
  const float* g = gate + b * DIM + c;
  v.x *= (1.0f + g[0]);
  v.y *= (1.0f + g[1]);
  v.z *= (1.0f + g[2]);
  v.w *= (1.0f + g[3]);
  o4[idx4] = v;
}

// ---------------- launch ----------------
extern "C" void kernel_launch(void* const* d_in, const int* in_sizes, int n_in,
                              void* d_out, int out_size, void* d_ws, size_t ws_size,
                              hipStream_t stream) {
  const float* x   = (const float*)d_in[0];
  const float* W1  = (const float*)d_in[1];
  const float* b1  = (const float*)d_in[2];
  const float* W2  = (const float*)d_in[3];
  const float* b2  = (const float*)d_in[4];
  const float* cw1 = (const float*)d_in[5];
  const float* cb1 = (const float*)d_in[6];
  const float* cw2 = (const float*)d_in[7];
  const float* cb2 = (const float*)d_in[8];
  float* out = (float*)d_out;

  char* ws = (char*)d_ws;
  unsigned short* w1t = (unsigned short*)ws;
  unsigned short* w2t = (unsigned short*)(ws + W1T_BYTES);
  float* pooled = (float*)(ws + POOLED_OFF);
  float* gate   = (float*)(ws + GATE_OFF);

  // phase 0: bf16 weight transposes + zero pooled accumulator
  convert_w1_kernel<<<(NUM_HEADS * HEAD_DIM * HIDDEN + 255) / 256, 256, 0, stream>>>(W1, w1t);
  convert_w2_kernel<<<(NUM_HEADS * HIDDEN * HEAD_DIM + 255) / 256, 256, 0, stream>>>(W2, w2t);
  zero_f32_kernel<<<(BATCH * DIM + 255) / 256, 256, 0, stream>>>(pooled, BATCH * DIM);

  // phase 1: fused per-head MLP (bf16 WMMA) + pooled sums
  dim3 grid1(BATCH * NTOK / TOK_TILE, NUM_HEADS);
  size_t smem = XS_BYTES + HS_BYTES;   // ~83 KB dynamic LDS (CDNA5: 320 KB/WGP)
  mlp_wmma_kernel<<<grid1, 256, smem, stream>>>(x, b1, b2, w1t, w2t, out, pooled);

  // phase 2: SE gate
  gate_kernel<<<BATCH, 256, 0, stream>>>(pooled, cw1, cb1, cw2, cb2, gate);

  // phase 3: out *= (1 + gate)
  apply_gate_kernel<<<BATCH * NTOK * DIM / 4 / 256, 256, 0, stream>>>(out, gate);
}
